// ToxicityLSTM_35759897706880
// MI455X (gfx1250) — compile-verified, matrix operand order checked by reading, not run on hardware
//
#include <hip/hip_runtime.h>

// ---------------------------------------------------------------------------
// 2-layer LSTM (B=256,T=512,E=H=256) + dense-sigmoid-threshold head, MI455X.
// Persistent workgroup per 16-row batch tile (batch rows are independent
// through the recurrence); bf16 weights pre-swizzled into WMMA B-fragment
// layout; v_wmma_f32_16x16x32_bf16 with f32 accumulation; activations, cell
// state, z-buffer, biases, head weights all resident in LDS (~144 KB / WGP).
// ---------------------------------------------------------------------------

typedef __attribute__((ext_vector_type(16))) __bf16 v16bf;
typedef __attribute__((ext_vector_type(8)))  float  v8f;

#define B_   256
#define T_   512
#define E_   256
#define H_   256
#define NZ   1024   // 4*H gate width
#define KDIM 512    // E+H (== H+H) reduction depth
#define C_   6

__device__ __forceinline__ unsigned short f2bf(float f) {
  unsigned u = __float_as_uint(f);
  u += 0x7FFFu + ((u >> 16) & 1u);          // round-to-nearest-even
  return (unsigned short)(u >> 16);
}
__device__ __forceinline__ float bf2f(unsigned short h) {
  return __uint_as_float(((unsigned)h) << 16);
}
__device__ __forceinline__ float sigmoidf_(float x) {
  return 1.0f / (1.0f + expf(-x));
}

// ---------------------------------------------------------------------------
// Pre-swizzle W1/W2 (fp32 [512,1024] row-major) into bf16 WMMA-B tiles.
// Tile (kt,nt) covers K=kt*32..+31, N=nt*16..+15; within a tile, lane l's 16
// halves are stored contiguously in the ISA's 16-bit B-fragment order:
//   lanes 0-15 : col N=l,     K = {0..7, 16..23}
//   lanes 16-31: col N=l-16,  K = {8..15, 24..31}
// so the main loop loads each B fragment as 32 contiguous bytes per lane.
// ---------------------------------------------------------------------------
__global__ void swizzle_weights(const float* __restrict__ W1,
                                const float* __restrict__ W2,
                                unsigned short* __restrict__ dst) {
  const int mat  = blockIdx.x >> 10;        // 0: W1, 1: W2
  const int tile = blockIdx.x & 1023;       // 16 kt * 64 nt
  const int kt   = tile >> 6;
  const int nt   = tile & 63;
  const float* src = mat ? W2 : W1;
  unsigned short* out = dst + (size_t)mat * (KDIM * NZ) + (size_t)tile * 512;
#pragma unroll
  for (int r = 0; r < 2; ++r) {
    const int h = threadIdx.x * 2 + r;      // 0..511 within tile
    const int l = h >> 4;                   // lane
    const int e = h & 15;                   // element within fragment
    const int kb = (l >= 16) ? 8 : 0;
    const int k = kt * 32 + ((e < 8) ? (kb + e) : (16 + kb + (e - 8)));
    const int n = nt * 16 + (l & 15);
    out[h] = f2bf(src[(size_t)k * NZ + n]);
  }
}

// ---------------------------------------------------------------------------
// One GEMM slice: z[16 x (wave's 64 cols)] = act[16 x 512] @ Wq, via WMMA.
// A fragment per ISA 16-bit A layout: lane<16 row M=lane, K={0..7,16..23};
// lane>=16 row M=lane-16, K={8..15,24..31}  -> two ds_load_b128 per fragment.
// ---------------------------------------------------------------------------
__device__ __forceinline__ void wmma_layer(const unsigned short* __restrict__ act,
                                           const unsigned short* __restrict__ wq,
                                           float* __restrict__ zbuf,
                                           int lane, int wave) {
  v8f acc[4];
#pragma unroll
  for (int nt = 0; nt < 4; ++nt)
#pragma unroll
    for (int e = 0; e < 8; ++e) acc[nt][e] = 0.0f;

  const int row  = lane & 15;
  const int koff = (lane >> 4) * 8;

  for (int kt = 0; kt < 16; ++kt) {
    union { uint4 u[2]; v16bf v; } A;
    const uint4* ap = (const uint4*)(act + row * KDIM + kt * 32 + koff);
    A.u[0] = ap[0];            // K = koff .. koff+7
    A.u[1] = ap[2];            // K = koff+16 .. koff+23
#pragma unroll
    for (int nt = 0; nt < 4; ++nt) {
      const int nG = wave * 4 + nt;                 // global 16-col tile index
      union { uint4 u[2]; v16bf v; } Bv;
      const uint4* bp =
          (const uint4*)(wq + ((size_t)(kt * 64 + nG)) * 512 + lane * 16);
      Bv.u[0] = bp[0];
      Bv.u[1] = bp[1];
      acc[nt] = __builtin_amdgcn_wmma_f32_16x16x32_bf16(
          false, A.v, false, Bv.v, (short)0, acc[nt], false, false);
    }
  }

  // C/D layout: VGPR e, lane L -> M = e + 8*(L>=16), N = (L&15) + 16*nG
  const int m0 = (lane >> 4) * 8;
  const int n0 = lane & 15;
#pragma unroll
  for (int nt = 0; nt < 4; ++nt) {
    const int n = (wave * 4 + nt) * 16 + n0;
#pragma unroll
    for (int e = 0; e < 8; ++e)
      zbuf[(m0 + e) * NZ + n] = acc[nt][e];
  }
}

// ---------------------------------------------------------------------------
// Elementwise gates: c' = c*sig(f+bias_f') + sig(i)*tanh(j); h' = tanh(c')*sig(o)
// (FORGET_BIAS already folded into the LDS-resident f-gate bias.)
// ---------------------------------------------------------------------------
__device__ __forceinline__ void gates(const float* __restrict__ zbuf,
                                      float* __restrict__ cst,
                                      const float* __restrict__ biasL,  // LDS
                                      unsigned short* __restrict__ act1,
                                      unsigned short* __restrict__ act2,
                                      int tid, int layer) {
  const int row = tid >> 5;               // 0..15
  const int cb  = (tid & 31) * 8;         // column base within H
#pragma unroll
  for (int q = 0; q < 8; ++q) {
    const int hc = cb + q;
    const float zi = zbuf[row * NZ + hc]          + biasL[hc];
    const float zj = zbuf[row * NZ + H_ + hc]     + biasL[H_ + hc];
    const float zf = zbuf[row * NZ + 2 * H_ + hc] + biasL[2 * H_ + hc];
    const float zo = zbuf[row * NZ + 3 * H_ + hc] + biasL[3 * H_ + hc];
    const float c  = cst[row * H_ + hc];
    const float cn = c * sigmoidf_(zf) + sigmoidf_(zi) * tanhf(zj);
    const float hn = tanhf(cn) * sigmoidf_(zo);
    cst[row * H_ + hc] = cn;
    const unsigned short hb = f2bf(hn);
    if (layer == 0) {
      act1[row * KDIM + H_ + hc] = hb;    // h1 -> layer-1 input (next step)
      act2[row * KDIM + hc]      = hb;    // h1 -> layer-2 input (this step)
    } else {
      act2[row * KDIM + H_ + hc] = hb;    // h2 -> layer-2 input + head
    }
  }
}

// ---------------------------------------------------------------------------
// Persistent LSTM kernel: 1 workgroup = 16 batch rows, 512 threads = 16 waves,
// each wave owns a 64-column slice of the 1024-wide gate GEMM.
// ---------------------------------------------------------------------------
__global__ void __launch_bounds__(512)
lstm_persistent(const float* __restrict__ x,
                const float* __restrict__ b1,
                const float* __restrict__ b2,
                const float* __restrict__ Wout,
                const float* __restrict__ bout,
                const unsigned short* __restrict__ wq,
                float* __restrict__ out_logits,
                float* __restrict__ out_cond,
                float* __restrict__ out_pred) {
  extern __shared__ unsigned char smem_raw[];
  unsigned short* act1 = (unsigned short*)smem_raw;        // 16x512 bf16: [x_t | h1]
  unsigned short* act2 = act1 + 16 * KDIM;                 // 16x512 bf16: [h1 | h2]
  float* zbuf  = (float*)(act2 + 16 * KDIM);               // 16x1024 f32
  float* c1    = zbuf + 16 * NZ;                           // 16x256 f32
  float* c2    = c1 + 16 * H_;                             // 16x256 f32
  float* bias1 = c2 + 16 * H_;                             // 1024 f32 (f-bias +1)
  float* bias2 = bias1 + NZ;                               // 1024 f32
  float* woutL = bias2 + NZ;                               // 256x6 f32
  float* boutL = woutL + H_ * C_;                          // 6 f32
  float* htmp  = boutL + 8;                                // 384 f32 head partials

  const int tid  = threadIdx.x;
  const int lane = tid & 31;
  const int wave = tid >> 5;                 // 0..15
  const int b0   = blockIdx.x * 16;

  const unsigned short* w1q = wq;
  const unsigned short* w2q = wq + (size_t)KDIM * NZ;

  // --- one-time init: states, LDS-resident biases / head weights -----------
  for (int i = tid; i < 16 * H_; i += blockDim.x) {
    const int r = i >> 8, c = i & 255;
    c1[i] = 0.0f;
    c2[i] = 0.0f;
    act1[r * KDIM + H_ + c] = 0;
    act2[r * KDIM + c]      = 0;
    act2[r * KDIM + H_ + c] = 0;
  }
  for (int i = tid; i < NZ; i += blockDim.x) {
    const float fb = (i >= 2 * H_ && i < 3 * H_) ? 1.0f : 0.0f;  // FORGET_BIAS
    bias1[i] = b1[i] + fb;
    bias2[i] = b2[i] + fb;
  }
  for (int i = tid; i < H_ * C_; i += blockDim.x) woutL[i] = Wout[i];
  if (tid < C_) boutL[tid] = bout[tid];
  __syncthreads();

  // head work assignment (constant over t): 384 threads x 64-element partials
  const int ho   = tid % 96;                 // output id: row*6+col
  const int hsub = tid / 96;                 // quarter of the H reduction
  const int hrow = ho / C_;
  const int hcol = ho % C_;

  for (int t = 0; t < T_; ++t) {
    // Phase 0: x_t -> act1[:, 0:256) as bf16 (vectorized fp32 loads)
    {
      const int row = tid >> 5;
      const int cb  = (tid & 31) * 8;
      const float* xp = x + ((size_t)(b0 + row) * T_ + t) * E_ + cb;
      const float4 a = ((const float4*)xp)[0];
      const float4 b = ((const float4*)xp)[1];
      __align__(16) unsigned short h[8] = {
          f2bf(a.x), f2bf(a.y), f2bf(a.z), f2bf(a.w),
          f2bf(b.x), f2bf(b.y), f2bf(b.z), f2bf(b.w)};
      *(uint4*)(&act1[row * KDIM + cb]) = *(const uint4*)h;
    }
    __syncthreads();

    // Phase 1: z = [x_t|h1] @ W1
    wmma_layer(act1, w1q, zbuf, lane, wave);
    __syncthreads();

    // Phase 2: layer-1 gates
    gates(zbuf, c1, bias1, act1, act2, tid, 0);
    __syncthreads();

    // Phase 3: z = [h1|h2] @ W2
    wmma_layer(act2, w2q, zbuf, lane, wave);
    __syncthreads();

    // Phase 4: layer-2 gates
    gates(zbuf, c2, bias2, act1, act2, tid, 1);
    __syncthreads();

    // Phase 5a: head partial dots (384 threads, 64 elements each, all LDS)
    if (tid < 384) {
      const int k0 = hsub * 64;
      float p = 0.0f;
#pragma unroll 8
      for (int k = 0; k < 64; ++k)
        p += bf2f(act2[hrow * KDIM + H_ + k0 + k]) * woutL[(k0 + k) * C_ + hcol];
      htmp[tid] = p;
    }
    __syncthreads();

    // Phase 5b: reduce + activation + threshold + store
    if (tid < 96) {
      const float acc = boutL[hcol] + htmp[tid] + htmp[96 + tid] +
                        htmp[192 + tid] + htmp[288 + tid];
      const size_t idx = ((size_t)(b0 + hrow) * T_ + t) * C_ + hcol;
      out_logits[idx] = acc;
      const float cond = (acc > 0.0f) ? 1.0f : 0.0f;  // sig(z)>0.5 <=> z>0
      out_cond[idx] = cond;
      out_pred[idx] = cond;
    }
    __syncthreads();  // protect htmp / act1 / zbuf before next iteration
  }
}

// ---------------------------------------------------------------------------
extern "C" void kernel_launch(void* const* d_in, const int* in_sizes, int n_in,
                              void* d_out, int out_size, void* d_ws, size_t ws_size,
                              hipStream_t stream) {
  const float* x    = (const float*)d_in[0];
  const float* W1   = (const float*)d_in[1];
  const float* b1   = (const float*)d_in[2];
  const float* W2   = (const float*)d_in[3];
  const float* b2   = (const float*)d_in[4];
  const float* Wout = (const float*)d_in[5];
  const float* bout = (const float*)d_in[6];
  unsigned short* wq = (unsigned short*)d_ws;     // 2 MB bf16 swizzled weights
  float* out = (float*)d_out;
  const int n1 = out_size / 3;                    // logits | condition | prediction

  swizzle_weights<<<2048, 256, 0, stream>>>(W1, W2, wq);

  const size_t shmem = (size_t)16 * KDIM * 2 * 2       // act1 + act2 (bf16)
                     + (size_t)16 * NZ * 4             // zbuf (f32)
                     + (size_t)16 * H_ * 4 * 2         // c1 + c2 (f32)
                     + (size_t)NZ * 4 * 2              // bias1 + bias2 (f32)
                     + (size_t)(H_ * C_ + 8 + 384) * 4;// Wout + bout + head tmp
  lstm_persistent<<<B_ / 16, 512, shmem, stream>>>(
      x, b1, b2, Wout, bout, wq, out, out + n1, out + 2 * (size_t)n1);
}
